// ImageCaptioningModel_5506148073616
// MI455X (gfx1250) — matrix operand order repaired and checked by reading, not compiled, not run on hardware
//
#include <hip/hip_runtime.h>
#include <hip/hip_bf16.h>
#include <math.h>

// ---------------------------------------------------------------------------
// Model dims
// ---------------------------------------------------------------------------
#define BB 64
#define TT 20
#define VV 32000
#define EE 256
#define HH 512
#define EPSBN 1e-5f

typedef __bf16 bf16_t;
typedef __bf16 v16bf __attribute__((ext_vector_type(16)));
typedef float  v8f   __attribute__((ext_vector_type(8)));

#define EPI_NONE      0
#define EPI_BIAS      1
#define EPI_BIAS_ACC  2
#define EPI_BN_RELU   3

union Frag { uint4 q[2]; v16bf v; };

__device__ __forceinline__ void load_frags(const bf16_t* __restrict__ a,
                                           const bf16_t* __restrict__ bk,
                                           Frag& fa, Frag& fb0, Frag& fb1,
                                           Frag& fb2, Frag& fb3)
{
    fa.q[0]  = *(const uint4*)(a);
    fa.q[1]  = *(const uint4*)(a + 8);
    fb0.q[0] = *(const uint4*)(bk);
    fb0.q[1] = *(const uint4*)(bk + 8);
    fb1.q[0] = *(const uint4*)(bk + 512);
    fb1.q[1] = *(const uint4*)(bk + 520);
    fb2.q[0] = *(const uint4*)(bk + 1024);
    fb2.q[1] = *(const uint4*)(bk + 1032);
    fb3.q[0] = *(const uint4*)(bk + 1536);
    fb3.q[1] = *(const uint4*)(bk + 1544);
}

// ---------------------------------------------------------------------------
// bf16 WMMA GEMM:  C(MxN, f32) = A(MxK bf16 row-major) * Bpack(K,N bf16, WMMA-blocked)
// One wave computes a 16x64 strip (4 N-tiles), reusing the A fragment, with
// software-pipelined (double-buffered) fragment loads so WMMAs overlap the
// 10 in-flight b128 loads of the next K-step.
// K % 32 == 0, M % 16 == 0, N % 64 == 0.
//
// Fragment layouts used here (wave32, contiguous k, not the ISA table order), g = lane/16:
//   A 16x32 : lane -> row M=lane%16 ; av elem e = A[row][kk + g*16 + e]
//             (two contiguous b128 chunks)
//   B 32x16 : lane -> col N=lane%16 ; bv elem e -> k = kt*32 + g*16 + e.
//             Bpack block (kt,nt): Bpack[(kt*ntiles + nt)*512 + lane*16 + e]
//   C 16x16 : lane -> col N=lane%16 ; VGPR r -> row M = g*8 + r
// ---------------------------------------------------------------------------
__global__ void gemm_bf16_wmma(const bf16_t* __restrict__ A, int lda,
                               const bf16_t* __restrict__ Bp, int ntilesN,
                               float* __restrict__ C, long long ldc,
                               int M, int N, int K,
                               const float* __restrict__ p0,
                               const float* __restrict__ p1,
                               int epi)
{
    const int wave    = blockIdx.x * 4 + (threadIdx.x >> 5);
    const int tilesN4 = N >> 6;                 // strips of 4 tiles
    const int tm      = wave / tilesN4;
    if (tm >= (M >> 4)) return;                 // wave-uniform guard (EXEC all-1)
    const int tn   = (wave % tilesN4) << 2;     // first of 4 n-tiles
    const int lane = threadIdx.x & 31;
    const int g    = lane >> 4;
    const int mr   = lane & 15;

    const bf16_t* arow  = A + (size_t)(tm * 16 + mr) * (size_t)lda + g * 16;
    const bf16_t* bbase = Bp + ((size_t)tn << 9) + lane * 16;
    const size_t  bstep = (size_t)ntilesN << 9; // elems between successive kt blocks

    v8f acc0 = {}, acc1 = {}, acc2 = {}, acc3 = {};

    Frag ca, cb0, cb1, cb2, cb3;                // current K-step fragments
    load_frags(arow, bbase, ca, cb0, cb1, cb2, cb3);

    for (int kk = 32; kk < K; kk += 32) {
        Frag na, nb0, nb1, nb2, nb3;            // prefetch next K-step into regs
        const bf16_t* bk = bbase + (size_t)(kk >> 5) * bstep;
        load_frags(arow + kk, bk, na, nb0, nb1, nb2, nb3);
        if (kk + 32 < K) {
            __builtin_prefetch(arow + kk + 32, 0, 3);      // global_prefetch_b8
            __builtin_prefetch(bk + bstep, 0, 3);
        }
        acc0 = __builtin_amdgcn_wmma_f32_16x16x32_bf16(false, ca.v, false, cb0.v,
                                                       (short)0, acc0, false, false);
        acc1 = __builtin_amdgcn_wmma_f32_16x16x32_bf16(false, ca.v, false, cb1.v,
                                                       (short)0, acc1, false, false);
        acc2 = __builtin_amdgcn_wmma_f32_16x16x32_bf16(false, ca.v, false, cb2.v,
                                                       (short)0, acc2, false, false);
        acc3 = __builtin_amdgcn_wmma_f32_16x16x32_bf16(false, ca.v, false, cb3.v,
                                                       (short)0, acc3, false, false);
        ca = na; cb0 = nb0; cb1 = nb1; cb2 = nb2; cb3 = nb3;
    }
    acc0 = __builtin_amdgcn_wmma_f32_16x16x32_bf16(false, ca.v, false, cb0.v,
                                                   (short)0, acc0, false, false);
    acc1 = __builtin_amdgcn_wmma_f32_16x16x32_bf16(false, ca.v, false, cb1.v,
                                                   (short)0, acc1, false, false);
    acc2 = __builtin_amdgcn_wmma_f32_16x16x32_bf16(false, ca.v, false, cb2.v,
                                                   (short)0, acc2, false, false);
    acc3 = __builtin_amdgcn_wmma_f32_16x16x32_bf16(false, ca.v, false, cb3.v,
                                                   (short)0, acc3, false, false);

    v8f accs[4] = {acc0, acc1, acc2, acc3};
#pragma unroll
    for (int t = 0; t < 4; ++t) {
        const int n = (tn + t) * 16 + mr;
#pragma unroll
        for (int r = 0; r < 8; ++r) {
            const int m = tm * 16 + g * 8 + r;
            float* cp = C + (size_t)m * (size_t)ldc + n;
            float  v  = accs[t][r];
            if (epi == EPI_BN_RELU)       v = fmaxf(v * p0[n] + p1[n], 0.f);
            else if (epi == EPI_BIAS)     v = v + p0[n];
            else if (epi == EPI_BIAS_ACC) v = *cp + v + p0[n];
            *cp = v;
        }
    }
}

// ---------------------------------------------------------------------------
// im2col (3x3, stride 2, pad 1) with generic input strides -> bf16 (rows,Kpad)
// ---------------------------------------------------------------------------
__global__ void im2col_kernel(const float* __restrict__ X,
                              int C_, int Hin, int Win,
                              size_t sb, size_t sc, size_t sy, size_t sx,
                              int OH, int OW, int Kpad,
                              bf16_t* __restrict__ out, size_t total)
{
    size_t idx = (size_t)blockIdx.x * blockDim.x + threadIdx.x;
    if (idx >= total) return;
    int    k   = (int)(idx % (size_t)Kpad);
    size_t row = idx / (size_t)Kpad;
    int ow = (int)(row % (size_t)OW);
    size_t t = row / (size_t)OW;
    int oh = (int)(t % (size_t)OH);
    int b  = (int)(t / (size_t)OH);
    float val = 0.f;
    if (k < C_ * 9) {
        int c = k / 9, rem = k % 9, ky = rem / 3, kx = rem % 3;
        int iy = oh * 2 - 1 + ky;
        int ix = ow * 2 - 1 + kx;
        if (iy >= 0 && iy < Hin && ix >= 0 && ix < Win)
            val = X[(size_t)b * sb + (size_t)c * sc + (size_t)iy * sy + (size_t)ix * sx];
    }
    out[idx] = (bf16_t)val;
}

// ---------------------------------------------------------------------------
// Weight packers producing the WMMA-blocked B layout:
//   out[(kt*(N/16)+nt)*512 + lane*16 + e] = B[k][n],
//   k = kt*32 + (lane/16)*16 + e ; n = nt*16 + (lane&15)
// ---------------------------------------------------------------------------
__global__ void pack_wT_wmma(const float* __restrict__ in, int N, int K,
                             bf16_t* __restrict__ out)     // in: (N,K) row-major
{
    size_t idx = (size_t)blockIdx.x * blockDim.x + threadIdx.x;
    size_t total = (size_t)K * N;
    if (idx >= total) return;
    int e    = (int)(idx & 15);
    int lane = (int)((idx >> 4) & 31);
    size_t bi = idx >> 9;
    int ntiles = N >> 4;
    int nt = (int)(bi % (size_t)ntiles);
    int kt = (int)(bi / (size_t)ntiles);
    int k = kt * 32 + ((lane >> 4) << 4) + e;
    int n = nt * 16 + (lane & 15);
    out[idx] = (bf16_t)in[(size_t)n * K + k];
}

// conv weight (CO,CI,3,3) f32 -> WMMA-blocked (Kpad, CO), k = ci*9+ky*3+kx (zero-padded)
__global__ void pack_conv_w_wmma(const float* __restrict__ w, int CO, int CI, int Kpad,
                                 bf16_t* __restrict__ out)
{
    size_t idx = (size_t)blockIdx.x * blockDim.x + threadIdx.x;
    size_t total = (size_t)Kpad * CO;
    if (idx >= total) return;
    int e    = (int)(idx & 15);
    int lane = (int)((idx >> 4) & 31);
    size_t bi = idx >> 9;
    int ntiles = CO >> 4;
    int nt = (int)(bi % (size_t)ntiles);
    int kt = (int)(bi / (size_t)ntiles);
    int k  = kt * 32 + ((lane >> 4) << 4) + e;
    int co = nt * 16 + (lane & 15);
    float v = 0.f;
    if (k < CI * 9) {
        int c = k / 9, rem = k % 9;
        v = w[((size_t)co * CI + c) * 9 + rem];
    }
    out[idx] = (bf16_t)v;
}

__global__ void bn_prepare(const float* __restrict__ g, const float* __restrict__ beta,
                           const float* __restrict__ m, const float* __restrict__ v,
                           const float* __restrict__ convb, int C,
                           float* __restrict__ scale, float* __restrict__ shift)
{
    int i = blockIdx.x * blockDim.x + threadIdx.x;
    if (i >= C) return;
    float sc = g[i] * rsqrtf(v[i] + EPSBN);
    scale[i] = sc;
    shift[i] = beta[i] + (convb[i] - m[i]) * sc;
}

__global__ void f32_to_bf16_kernel(const float* __restrict__ in,
                                   bf16_t* __restrict__ out, size_t n)
{
    size_t i = (size_t)blockIdx.x * blockDim.x + threadIdx.x;
    if (i < n) out[i] = (bf16_t)in[i];
}

__global__ void fill_f32(float* p, float v, size_t n)
{
    size_t i = (size_t)blockIdx.x * blockDim.x + threadIdx.x;
    if (i < n) p[i] = v;
}
__global__ void fill_bf16(bf16_t* p, float v, size_t n)
{
    size_t i = (size_t)blockIdx.x * blockDim.x + threadIdx.x;
    if (i < n) p[i] = (bf16_t)v;
}

// emb (V,E) f32 + captions (B,T) int -> emb_seq (T,B,E) bf16
__global__ void embed_gather(const float* __restrict__ emb,
                             const int* __restrict__ captions,
                             bf16_t* __restrict__ out)
{
    size_t idx = (size_t)blockIdx.x * blockDim.x + threadIdx.x;
    size_t total = (size_t)TT * BB * EE;
    if (idx >= total) return;
    int e = (int)(idx % EE);
    int b = (int)((idx / EE) % BB);
    int t = (int)(idx / ((size_t)EE * BB));
    int tok = captions[b * TT + t];
    out[idx] = (bf16_t)emb[(size_t)tok * EE + e];
}

// scores[b,r] = sum_h tanh(v_proj[b,r,h] + tbuf[b,h]) * wa[h] + ba ; one wave per (b,r)
__global__ void att_scores_kernel(const float* __restrict__ v_proj,
                                  const float* __restrict__ tbuf,
                                  const float* __restrict__ wa,
                                  const float* __restrict__ ba,
                                  float* __restrict__ scores)
{
    int pair = blockIdx.x * 4 + (threadIdx.x >> 5);
    if (pair >= BB * 16) return;
    int lane = threadIdx.x & 31;
    int b = pair >> 4;
    const float* vp = v_proj + (size_t)pair * HH;
    const float* tb = tbuf + (size_t)b * HH;
    float s = 0.f;
    for (int h = lane; h < HH; h += 32)
        s += tanhf(vp[h] + tb[h]) * wa[h];
#pragma unroll
    for (int off = 16; off; off >>= 1) s += __shfl_down(s, off, 32);
    if (lane == 0) scores[pair] = s + ba[0];
}

// per-b softmax over 16 regions, attended = sum_r w[r]*visual_e[b,r,:],
// xin = [word_embed | attended] in bf16
__global__ void att_attend_kernel(const float* __restrict__ scores,
                                  const float* __restrict__ visual_e,
                                  const bf16_t* __restrict__ wemb,   // (B,E) for this t
                                  bf16_t* __restrict__ xin)
{
    int b = blockIdx.x;
    __shared__ float w[16];
    int tid = threadIdx.x;
    if (tid < 16) w[tid] = scores[b * 16 + tid];
    __syncthreads();
    if (tid == 0) {
        float mx = w[0];
        for (int i = 1; i < 16; ++i) mx = fmaxf(mx, w[i]);
        float sum = 0.f;
        for (int i = 0; i < 16; ++i) { w[i] = expf(w[i] - mx); sum += w[i]; }
        float inv = 1.f / sum;
        for (int i = 0; i < 16; ++i) w[i] *= inv;
    }
    __syncthreads();
    float acc = 0.f;                                   // tid indexes E=256
#pragma unroll
    for (int r = 0; r < 16; ++r)
        acc += w[r] * visual_e[((size_t)b * 16 + r) * EE + tid];
    xin[(size_t)b * (2 * EE) + tid]      = wemb[(size_t)b * EE + tid];
    xin[(size_t)b * (2 * EE) + EE + tid] = (bf16_t)acc;
}

__device__ __forceinline__ float sigmoidf_(float x) { return 1.f / (1.f + expf(-x)); }

// gates (B,4H) f32 (torch order i,f,g,o) -> update c (f32) and h (bf16)
__global__ void lstm_cell_kernel(const float* __restrict__ gates,
                                 float* __restrict__ cstate,
                                 bf16_t* __restrict__ h_bf)
{
    int b = blockIdx.x;
    int j = threadIdx.x;                               // H = 512
    const float* gr = gates + (size_t)b * (4 * HH);
    float ig = sigmoidf_(gr[j]);
    float fg = sigmoidf_(gr[HH + j]);
    float gg = tanhf(gr[2 * HH + j]);
    float og = sigmoidf_(gr[3 * HH + j]);
    float c  = fg * cstate[(size_t)b * HH + j] + ig * gg;
    float h  = og * tanhf(c);
    cstate[(size_t)b * HH + j] = c;
    h_bf[(size_t)b * HH + j]   = (bf16_t)h;
}

// ---------------------------------------------------------------------------
// Host side
// ---------------------------------------------------------------------------
static inline void launch_gemm(const bf16_t* A, int lda, const bf16_t* Bp,
                               float* C, long long ldc, int M, int N, int K,
                               const float* p0, const float* p1, int epi,
                               hipStream_t s)
{
    int waves  = (M / 16) * (N / 64);
    int blocks = (waves + 3) / 4;
    gemm_bf16_wmma<<<blocks, 128, 0, s>>>(A, lda, Bp, N / 16, C, ldc, M, N, K,
                                          p0, p1, epi);
}

static inline int gsz(size_t total, int block) { return (int)((total + block - 1) / block); }

extern "C" void kernel_launch(void* const* d_in, const int* in_sizes, int n_in,
                              void* d_out, int out_size, void* d_ws, size_t ws_size,
                              hipStream_t stream)
{
    (void)in_sizes; (void)n_in; (void)out_size; (void)ws_size;

    const float* images   = (const float*)d_in[0];
    const int*   captions = (const int*)d_in[1];
    const float* convw[4] = {(const float*)d_in[2],  (const float*)d_in[8],
                             (const float*)d_in[14], (const float*)d_in[20]};
    const float* convb[4] = {(const float*)d_in[3],  (const float*)d_in[9],
                             (const float*)d_in[15], (const float*)d_in[21]};
    const float* bng[4]   = {(const float*)d_in[4],  (const float*)d_in[10],
                             (const float*)d_in[16], (const float*)d_in[22]};
    const float* bnb[4]   = {(const float*)d_in[5],  (const float*)d_in[11],
                             (const float*)d_in[17], (const float*)d_in[23]};
    const float* bnm[4]   = {(const float*)d_in[6],  (const float*)d_in[12],
                             (const float*)d_in[18], (const float*)d_in[24]};
    const float* bnv[4]   = {(const float*)d_in[7],  (const float*)d_in[13],
                             (const float*)d_in[19], (const float*)d_in[25]};
    const float* proj_w = (const float*)d_in[26];
    const float* proj_b = (const float*)d_in[27];
    const float* emb    = (const float*)d_in[28];
    const float* att_wv = (const float*)d_in[29];
    const float* att_bv = (const float*)d_in[30];
    const float* att_wt = (const float*)d_in[31];
    const float* att_bt = (const float*)d_in[32];
    const float* att_wa = (const float*)d_in[33];
    const float* att_ba = (const float*)d_in[34];
    const float* w_ih   = (const float*)d_in[35];
    const float* w_hh   = (const float*)d_in[36];
    const float* b_ih   = (const float*)d_in[37];
    const float* b_hh   = (const float*)d_in[38];
    const float* out_w  = (const float*)d_in[39];
    const float* out_b  = (const float*)d_in[40];
    float* out = (float*)d_out;

    // ---- workspace bump allocator ----
    char* wp = (char*)d_ws;
    auto alloc = [&](size_t bytes) -> void* {
        void* p = wp; wp += (bytes + 255) & ~(size_t)255; return p;
    };
    bf16_t* im2col   = (bf16_t*)alloc((size_t)16384 * 576 * 2);   // max im2col slab
    float*  actA     = (float*) alloc((size_t)65536 * 64 * 4);    // act1 / act3
    float*  actB     = (float*) alloc((size_t)16384 * 128 * 4);   // act2 / act4
    bf16_t* wpack    = (bf16_t*)alloc((size_t)2304 * 512 * 2);    // conv weight pack
    float*  bnscale  = (float*) alloc(512 * 4);
    float*  bnshift  = (float*) alloc(512 * 4);
    bf16_t* projWT   = (bf16_t*)alloc((size_t)512 * 256 * 2);
    bf16_t* attWvT   = (bf16_t*)alloc((size_t)256 * 512 * 2);
    bf16_t* attWtT   = (bf16_t*)alloc((size_t)512 * 512 * 2);
    bf16_t* wihT     = (bf16_t*)alloc((size_t)512 * 2048 * 2);
    bf16_t* whhT     = (bf16_t*)alloc((size_t)512 * 2048 * 2);
    bf16_t* outWT    = (bf16_t*)alloc((size_t)512 * (size_t)VV * 2);
    bf16_t* visualBF = (bf16_t*)alloc((size_t)1024 * 512 * 2);
    float*  visualE  = (float*) alloc((size_t)1024 * 256 * 4);
    bf16_t* visualEB = (bf16_t*)alloc((size_t)1024 * 256 * 2);
    float*  vproj    = (float*) alloc((size_t)1024 * 512 * 4);
    bf16_t* embSeq   = (bf16_t*)alloc((size_t)TT * BB * EE * 2);
    bf16_t* h_bf     = (bf16_t*)alloc((size_t)BB * HH * 2);
    float*  cstate   = (float*) alloc((size_t)BB * HH * 4);
    float*  tbuf     = (float*) alloc((size_t)BB * HH * 4);
    float*  scores   = (float*) alloc((size_t)BB * 16 * 4);
    bf16_t* xin      = (bf16_t*)alloc((size_t)BB * 2 * EE * 2);
    float*  gates    = (float*) alloc((size_t)BB * 4 * HH * 4);

    // ---- encoder: 4x (im2col -> WMMA GEMM with fused BN+ReLU) ----
    struct ConvCfg {
        const float* X; int C, Hin, Win, OH, OW, Kpad, CO;
        size_t sb, sc, sy, sx; float* Y;
    };
    ConvCfg cc[4] = {
        // conv1: NCHW images (64,3,64,64) -> act1 NHWC (64,32,32,64)
        { images, 3, 64, 64, 32, 32, 32, 64,
          (size_t)3 * 64 * 64, (size_t)64 * 64, 64, 1, actA },
        // conv2: act1 NHWC -> act2 NHWC (64,16,16,128)
        { actA, 64, 32, 32, 16, 16, 576, 128,
          (size_t)32 * 32 * 64, 1, (size_t)32 * 64, 64, actB },
        // conv3: act2 -> act3 NHWC (64,8,8,256)  (reuses actA)
        { actB, 128, 16, 16, 8, 8, 1152, 256,
          (size_t)16 * 16 * 128, 1, (size_t)16 * 128, 128, actA },
        // conv4: act3 -> act4 NHWC (64,4,4,512)  (reuses actB)
        { actA, 256, 8, 8, 4, 4, 2304, 512,
          (size_t)8 * 8 * 256, 1, (size_t)8 * 256, 256, actB },
    };
    for (int L = 0; L < 4; ++L) {
        const ConvCfg& c = cc[L];
        size_t rows  = (size_t)BB * c.OH * c.OW;
        size_t total = rows * (size_t)c.Kpad;
        im2col_kernel<<<gsz(total, 256), 256, 0, stream>>>(
            c.X, c.C, c.Hin, c.Win, c.sb, c.sc, c.sy, c.sx,
            c.OH, c.OW, c.Kpad, im2col, total);
        size_t wtot = (size_t)c.Kpad * c.CO;
        pack_conv_w_wmma<<<gsz(wtot, 256), 256, 0, stream>>>(
            convw[L], c.CO, c.C, c.Kpad, wpack);
        bn_prepare<<<gsz(c.CO, 128), 128, 0, stream>>>(
            bng[L], bnb[L], bnm[L], bnv[L], convb[L], c.CO, bnscale, bnshift);
        launch_gemm(im2col, c.Kpad, wpack, c.Y, c.CO,
                    (int)rows, c.CO, c.Kpad, bnscale, bnshift, EPI_BN_RELU, stream);
    }

    // ---- decoder weight packs (WMMA-blocked layout) ----
    pack_wT_wmma<<<gsz((size_t)256 * 512, 256), 256, 0, stream>>>(proj_w, 256, 512, projWT);
    pack_wT_wmma<<<gsz((size_t)512 * 256, 256), 256, 0, stream>>>(att_wv, 512, 256, attWvT);
    pack_wT_wmma<<<gsz((size_t)512 * 512, 256), 256, 0, stream>>>(att_wt, 512, 512, attWtT);
    pack_wT_wmma<<<gsz((size_t)2048 * 512, 256), 256, 0, stream>>>(w_ih, 2048, 512, wihT);
    pack_wT_wmma<<<gsz((size_t)2048 * 512, 256), 256, 0, stream>>>(w_hh, 2048, 512, whhT);
    pack_wT_wmma<<<gsz((size_t)VV * 512, 256), 256, 0, stream>>>(out_w, VV, 512, outWT);

    // ---- visual projection + attention precompute ----
    f32_to_bf16_kernel<<<gsz((size_t)1024 * 512, 256), 256, 0, stream>>>(
        actB, visualBF, (size_t)1024 * 512);
    launch_gemm(visualBF, 512, projWT, visualE, 256,
                1024, 256, 512, proj_b, nullptr, EPI_BIAS, stream);
    f32_to_bf16_kernel<<<gsz((size_t)1024 * 256, 256), 256, 0, stream>>>(
        visualE, visualEB, (size_t)1024 * 256);
    launch_gemm(visualEB, 256, attWvT, vproj, 512,
                1024, 512, 256, att_bv, nullptr, EPI_BIAS, stream);

    // ---- embeddings + state init ----
    embed_gather<<<gsz((size_t)TT * BB * EE, 256), 256, 0, stream>>>(emb, captions, embSeq);
    fill_bf16<<<gsz((size_t)BB * HH, 256), 256, 0, stream>>>(h_bf, 0.f, (size_t)BB * HH);
    fill_f32<<<gsz((size_t)BB * HH, 256), 256, 0, stream>>>(cstate, 0.f, (size_t)BB * HH);

    // ---- decoder loop (T=20 steps, all on stream) ----
    for (int t = 0; t < TT; ++t) {
        // t = h @ att_wt.T + att_bt
        launch_gemm(h_bf, HH, attWtT, tbuf, HH,
                    BB, HH, HH, att_bt, nullptr, EPI_BIAS, stream);
        // attention scores + softmax + attend, build xin = [wemb | attended]
        att_scores_kernel<<<gsz((size_t)BB * 16, 4), 128, 0, stream>>>(
            vproj, tbuf, att_wa, att_ba, scores);
        att_attend_kernel<<<BB, EE, 0, stream>>>(
            scores, visualE, embSeq + (size_t)t * BB * EE, xin);
        // gates = xin @ w_ih.T + b_ih + h @ w_hh.T + b_hh
        launch_gemm(xin, 2 * EE, wihT, gates, 4 * HH,
                    BB, 4 * HH, 2 * EE, b_ih, nullptr, EPI_BIAS, stream);
        launch_gemm(h_bf, HH, whhT, gates, 4 * HH,
                    BB, 4 * HH, HH, b_hh, nullptr, EPI_BIAS_ACC, stream);
        // LSTM cell
        lstm_cell_kernel<<<BB, HH, 0, stream>>>(gates, cstate, h_bf);
        // logits -> out[:, t, :]  (ldc = T*V so rows land at b*T*V + t*V)
        launch_gemm(h_bf, HH, outWT, out + (size_t)t * VV, (long long)TT * VV,
                    BB, VV, HH, out_b, nullptr, EPI_BIAS, stream);
    }
}
